// GraphCastEncoder_87814901334435
// MI455X (gfx1250) — matrix operand order repaired and checked by reading, not compiled, not run on hardware
//
#include <hip/hip_runtime.h>

// ---------------------------------------------------------------------------
// GraphCast encoder for MI455X (gfx1250): all GEMMs via v_wmma_f32_16x16x32_bf16
// 256-thread blocks (8 waves/WGP on the big edge kernel), LDS overlay of the
// f32 LayerNorm staging tile onto the dead A tile.
// ---------------------------------------------------------------------------

typedef __attribute__((ext_vector_type(16))) __bf16 v16bf;
typedef __attribute__((ext_vector_type(8)))  float  v8f;

#define HD 128   // hidden dim H
#define NWAVE 8  // waves per block
#define BLK (NWAVE * 32)

__device__ __forceinline__ float4 ld4(const float* p) {
  return *reinterpret_cast<const float4*>(p);
}
__device__ __forceinline__ void st4(float* p, float4 v) {
  *reinterpret_cast<float4*>(p) = v;
}
__device__ __forceinline__ float silu_f(float x) {
  return x * (1.0f / (1.0f + __expf(-x)));
}

// Per-wave scratch: Hs (bf16 16x128) + union(Aw bf16 16xDINP, Ow f32 16x128).
constexpr size_t perwave_bytes(int DINP) {
  size_t aw = (size_t)16 * DINP * 2;
  size_t ow = (size_t)16 * 128 * 4;
  size_t un = aw > ow ? aw : ow;
  return (size_t)16 * 128 * 2 + un;
}
constexpr size_t lds_bytes(int DINP) {
  return (size_t)128 * DINP * 2          // W1^T  (bf16)  [128 n][DINP k]
       + (size_t)128 * 128 * 2           // W2^T  (bf16)  [128 n][128 k]
       + (size_t)4 * 128 * 4             // b1,b2,gamma,beta (f32)
       + (size_t)NWAVE * perwave_bytes(DINP);
}

// MODE: 0 = plain MLP, 1 = residual MLP, 2 = concat2(in0,in1)+residual(in0),
//       3 = edge: concat(g[src], m[dst], e) + residual(e) + atomic segment-sum
template<int DIN, int DINP, int MODE>
__global__ __launch_bounds__(BLK) void fused_mlp_kernel(
    const float* __restrict__ in0, const float* __restrict__ in1,
    const float* __restrict__ e_src, const long long* __restrict__ eidx,
    const float* __restrict__ resid, float* __restrict__ agg,
    float* __restrict__ out,
    const float* __restrict__ W1, const float* __restrict__ B1,
    const float* __restrict__ W2, const float* __restrict__ B2,
    const float* __restrict__ GAM, const float* __restrict__ BET,
    int nrows)
{
  extern __shared__ char smem[];
  __bf16* Wt1 = (__bf16*)smem;                 // [128][DINP] = W1 transposed
  __bf16* Wt2 = Wt1 + 128 * DINP;              // [128][128]  = W2 transposed
  float*  pb1 = (float*)(Wt2 + 128 * 128);
  float*  pb2 = pb1 + 128;
  float*  pg  = pb2 + 128;
  float*  pbt = pg  + 128;
  char*   wbase = (char*)(pbt + 128);

  const int wave = threadIdx.x >> 5;
  const int lane = threadIdx.x & 31;
  char* mybase = wbase + (size_t)wave * perwave_bytes(DINP);
  __bf16* Hs = (__bf16*)mybase;                // [16][128] hidden (bf16)
  __bf16* Aw = (__bf16*)(mybase + 16 * 128 * 2); // [16][DINP] A tile (bf16)
  float*  Ow = (float*) (mybase + 16 * 128 * 2); // overlays Aw after GEMM1

  // ---- cooperative weight staging: transpose + f32->bf16 ----
  for (int i = threadIdx.x; i < 128 * DINP; i += BLK) {
    int n = i / DINP, k = i % DINP;
    Wt1[i] = (k < DIN) ? (__bf16)W1[(size_t)k * 128 + n] : (__bf16)0.0f;
  }
  for (int i = threadIdx.x; i < 128 * 128; i += BLK) {
    int n = i >> 7, k = i & 127;
    Wt2[i] = (__bf16)W2[(size_t)k * 128 + n];
  }
  if (threadIdx.x < 128) {
    pb1[threadIdx.x] = B1[threadIdx.x];
    pb2[threadIdx.x] = B2[threadIdx.x];
    pg [threadIdx.x] = GAM[threadIdx.x];
    pbt[threadIdx.x] = BET[threadIdx.x];
  }
  __syncthreads();

  const long long row0 = ((long long)blockIdx.x * NWAVE + wave) * 16;
  if (row0 >= nrows) return;

  // ---- stage A tile: 16 rows x DINP cols, bf16, zero-padded ----
  if constexpr (DIN % 4 == 0) {
    constexpr int C4 = DINP / 4;
    for (int idx = lane; idx < 16 * C4; idx += 32) {
      int r = idx / C4, col = (idx % C4) * 4;
      long long grow = row0 + r;
      float4 v = make_float4(0.f, 0.f, 0.f, 0.f);
      if (grow < nrows) {
        if constexpr (MODE == 3) {
          if (col < 128)      { long long s = eidx[grow];         v = ld4(&in0[s * HD + col]); }
          else if (col < 256) { long long d = eidx[nrows + grow]; v = ld4(&in1[d * HD + (col - 128)]); }
          else                {                                   v = ld4(&e_src[grow * HD + (col - 256)]); }
        } else if constexpr (MODE == 2) {
          v = (col < 128) ? ld4(&in0[grow * HD + col])
                          : ld4(&in1[grow * HD + (col - 128)]);
        } else {
          if (col < DIN) v = ld4(&in0[grow * (long long)DIN + col]);
        }
      }
      __bf16* ap = &Aw[r * DINP + col];
      ap[0] = (__bf16)v.x; ap[1] = (__bf16)v.y;
      ap[2] = (__bf16)v.z; ap[3] = (__bf16)v.w;
    }
  } else {
    for (int idx = lane; idx < 16 * DINP; idx += 32) {
      int r = idx / DINP, col = idx % DINP;
      long long grow = row0 + r;
      float v = (grow < nrows && col < DIN) ? in0[grow * (long long)DIN + col] : 0.0f;
      Aw[idx] = (__bf16)v;
    }
  }
  asm volatile("s_wait_dscnt 0" ::: "memory");

  // WMMA fragment geometry (wave32; contiguous k order chosen by this code for both A and B):
  //   A 16x32 bf16: lane = (K-group<<4)|M ; v16bf = 16 contiguous K
  //   B 32x16 bf16 (from transposed LDS): lane = (K-group<<4)|N ; 16 contiguous K
  //   C/D f32: VGPR v -> M = v + 8*(lane>=16), N = 16*nt + (lane&15)
  const int rA   = lane & 15;
  const int kgA  = (lane >> 4) << 4;   // 0 or 16
  const int nB   = lane & 15;
  const int kgB  = (lane >> 4) << 4;   // 0 or 16
  const int Mrow = (lane >> 4) << 3;   // 0 or 8

  // ---- GEMM1: [16 x DINP] x W1 -> [16 x 128] f32 ----
  v8f acc[8];
  #pragma unroll
  for (int nt = 0; nt < 8; ++nt)
    #pragma unroll
    for (int el = 0; el < 8; ++el) acc[nt][el] = 0.0f;

  for (int k0 = 0; k0 < DINP; k0 += 32) {
    v16bf a;
    const __bf16* ab = &Aw[rA * DINP + k0 + kgA];
    #pragma unroll
    for (int j = 0; j < 8; ++j) { a[j] = ab[j]; a[8 + j] = ab[8 + j]; }
    #pragma unroll
    for (int nt = 0; nt < 8; ++nt) {
      const __bf16* bb = &Wt1[(nt * 16 + nB) * DINP + k0 + kgB];
      v16bf b;
      #pragma unroll
      for (int t = 0; t < 16; ++t) b[t] = bb[t];
      acc[nt] = __builtin_amdgcn_wmma_f32_16x16x32_bf16(
          false, a, false, b, (short)0, acc[nt], false, false);
    }
  }

  // ---- bias + SiLU -> hidden tile (bf16) ----
  #pragma unroll
  for (int nt = 0; nt < 8; ++nt) {
    int col = nt * 16 + (lane & 15);
    float bias = pb1[col];
    #pragma unroll
    for (int el = 0; el < 8; ++el) {
      int M = Mrow + el;
      Hs[M * HD + col] = (__bf16)silu_f(acc[nt][el] + bias);
    }
  }
  asm volatile("s_wait_dscnt 0" ::: "memory");

  // ---- GEMM2: [16 x 128] x W2 -> [16 x 128] f32 ----
  v8f acc2[8];
  #pragma unroll
  for (int nt = 0; nt < 8; ++nt)
    #pragma unroll
    for (int el = 0; el < 8; ++el) acc2[nt][el] = 0.0f;

  #pragma unroll
  for (int k0 = 0; k0 < 128; k0 += 32) {
    v16bf a;
    const __bf16* ab = &Hs[rA * HD + k0 + kgA];
    #pragma unroll
    for (int j = 0; j < 8; ++j) { a[j] = ab[j]; a[8 + j] = ab[8 + j]; }
    #pragma unroll
    for (int nt = 0; nt < 8; ++nt) {
      const __bf16* bb = &Wt2[(nt * 16 + nB) * HD + k0 + kgB];
      v16bf b;
      #pragma unroll
      for (int t = 0; t < 16; ++t) b[t] = bb[t];
      acc2[nt] = __builtin_amdgcn_wmma_f32_16x16x32_bf16(
          false, a, false, b, (short)0, acc2[nt], false, false);
    }
  }

  // ---- bias -> Ow (f32, overlays dead A tile) ----
  #pragma unroll
  for (int nt = 0; nt < 8; ++nt) {
    int col = nt * 16 + (lane & 15);
    float bias = pb2[col];
    #pragma unroll
    for (int el = 0; el < 8; ++el) {
      int M = Mrow + el;
      Ow[M * HD + col] = acc2[nt][el] + bias;
    }
  }
  asm volatile("s_wait_dscnt 0" ::: "memory");

  // ---- LayerNorm (+ residual, + atomic segment-sum for MODE==3) ----
  // lane pair (l, l^16) splits one 128-wide row; reduce with shfl_xor(16).
  {
    const int r     = lane & 15;
    const int halfw = lane >> 4;
    const long long grow = row0 + r;
    const float* orow = &Ow[r * HD + halfw * 64];
    float s = 0.0f, s2 = 0.0f;
    #pragma unroll
    for (int i = 0; i < 16; ++i) {
      float4 v = ((const float4*)orow)[i];
      s  += v.x + v.y + v.z + v.w;
      s2 += v.x * v.x + v.y * v.y + v.z * v.z + v.w * v.w;
    }
    s  += __shfl_xor(s,  16, 32);
    s2 += __shfl_xor(s2, 16, 32);
    const float mean = s * (1.0f / 128.0f);
    const float rstd = rsqrtf(s2 * (1.0f / 128.0f) - mean * mean + 1e-5f);

    if (grow < nrows) {
      long long dsti = 0;
      if constexpr (MODE == 3) dsti = eidx[nrows + grow];
      #pragma unroll
      for (int i = 0; i < 16; ++i) {
        int col = halfw * 64 + i * 4;
        float4 v = ((const float4*)orow)[i];
        float4 o;
        o.x = (v.x - mean) * rstd * pg[col + 0] + pbt[col + 0];
        o.y = (v.y - mean) * rstd * pg[col + 1] + pbt[col + 1];
        o.z = (v.z - mean) * rstd * pg[col + 2] + pbt[col + 2];
        o.w = (v.w - mean) * rstd * pg[col + 3] + pbt[col + 3];
        if constexpr (MODE == 3) {
          float* ap = &agg[dsti * HD + col];
          atomicAdd(ap + 0, o.x); atomicAdd(ap + 1, o.y);
          atomicAdd(ap + 2, o.z); atomicAdd(ap + 3, o.w);
        }
        if constexpr (MODE != 0) {
          float4 rv = ld4(&resid[grow * HD + col]);
          o.x += rv.x; o.y += rv.y; o.z += rv.z; o.w += rv.w;
        }
        st4(&out[grow * HD + col], o);
      }
    }
  }
}

// ---------------------------------------------------------------------------
extern "C" void kernel_launch(void* const* d_in, const int* in_sizes, int n_in,
                              void* d_out, int out_size, void* d_ws, size_t ws_size,
                              hipStream_t stream) {
  (void)in_sizes; (void)n_in; (void)out_size; (void)ws_size;
  constexpr int NG = 262144, NM = 40962, NE = 524288;

  const float*     grid_nfeat = (const float*)d_in[0];
  const float*     mesh_nfeat = (const float*)d_in[1];
  const long long* eidx       = (const long long*)d_in[2];
  const float*     efeat      = (const float*)d_in[3];
  // MLP param leaves, 6 per MLP, in dict order:
  const float* const* P = (const float* const*)(d_in + 4);
  // P[0..5]=grid_emb  P[6..11]=mesh_emb  P[12..17]=edge_emb
  // P[18..23]=ip_edge_mlp  P[24..29]=ip_node_mlp  P[30..35]=grid_node_mlp

  float* g_emb = (float*)d_ws;
  float* m_emb = g_emb + (size_t)NG * HD;
  float* e_emb = m_emb + (size_t)NM * HD;
  float* agg   = e_emb + (size_t)NE * HD;

  float* g_out = (float*)d_out;
  float* m_out = g_out + (size_t)NG * HD;
  float* e_out = m_out + (size_t)NM * HD;

  hipMemsetAsync(agg, 0, (size_t)NM * HD * sizeof(float), stream);

  auto nblk = [](int n) { return (unsigned)((n + (16 * NWAVE - 1)) / (16 * NWAVE)); };

  // 1) grid embedding: [NG,128] -> g_emb
  fused_mlp_kernel<128, 128, 0><<<nblk(NG), BLK, lds_bytes(128), stream>>>(
      grid_nfeat, nullptr, nullptr, nullptr, nullptr, nullptr, g_emb,
      P[0], P[1], P[2], P[3], P[4], P[5], NG);
  // 2) mesh embedding: [NM,3] -> m_emb (K padded 3->32)
  fused_mlp_kernel<3, 32, 0><<<nblk(NM), BLK, lds_bytes(32), stream>>>(
      mesh_nfeat, nullptr, nullptr, nullptr, nullptr, nullptr, m_emb,
      P[6], P[7], P[8], P[9], P[10], P[11], NM);
  // 3) edge embedding: [NE,4] -> e_emb (K padded 4->32)
  fused_mlp_kernel<4, 32, 0><<<nblk(NE), BLK, lds_bytes(32), stream>>>(
      efeat, nullptr, nullptr, nullptr, nullptr, nullptr, e_emb,
      P[12], P[13], P[14], P[15], P[16], P[17], NE);
  // 4) edge interaction: concat(g[src], m[dst], e) -> e_upd;
  //    e_out = e_emb + e_upd; agg[dst] += e_upd (atomic segment-sum)
  fused_mlp_kernel<384, 384, 3><<<nblk(NE), BLK, lds_bytes(384), stream>>>(
      g_emb, m_emb, e_emb, eidx, e_emb, agg, e_out,
      P[18], P[19], P[20], P[21], P[22], P[23], NE);
  // 5) mesh node update: concat(m_emb, agg) -> n_upd; m_out = m_emb + n_upd
  fused_mlp_kernel<256, 256, 2><<<nblk(NM), BLK, lds_bytes(256), stream>>>(
      m_emb, agg, nullptr, nullptr, m_emb, nullptr, m_out,
      P[24], P[25], P[26], P[27], P[28], P[29], NM);
  // 6) grid node residual MLP: g_out = g_emb + mlp(g_emb)
  fused_mlp_kernel<128, 128, 1><<<nblk(NG), BLK, lds_bytes(128), stream>>>(
      g_emb, nullptr, nullptr, nullptr, g_emb, nullptr, g_out,
      P[30], P[31], P[32], P[33], P[34], P[35], NG);
}
